// Service_25993142076017
// MI455X (gfx1250) — matrix operand order repaired and checked by reading, not compiled, not run on hardware
//
#include <hip/hip_runtime.h>

// ---------------------------------------------------------------------------
// Gather + GEMM:  out[b, :] = service_matrix[data[b], :] @ embedding_matrix
//   M = BATCH = 16384, K = N_FEATURES = 16384, N = EMBED_DIM = 64
//
// MI455X plan:
//   * bf16 WMMA (v_wmma_f32_16x16x32_bf16, f32 accumulate). Memory floor
//     (~360 MB gathered A @ 23.3 TB/s ~= 16 us) >> bf16 matrix time, so the
//     kernel is HBM-bound; all effort goes into keeping loads in flight
//     across the WMMAs.
//   * B (embedding, transposed to bf16 [64][16384] in d_ws) is staged into
//     LDS per 128-wide K-chunk with gfx1250 async global->LDS copies,
//     double-buffered across chunks, shared by all 8 waves of the block.
//   * Gathered A rows are register-pipelined one K-tile ahead (global), and
//     B fragments are register-pipelined one K-tile ahead (LDS), so neither
//     s_wait_loadcnt 0 nor s_wait_dscnt 0 sits directly in front of a WMMA.
// ---------------------------------------------------------------------------

typedef __attribute__((ext_vector_type(16))) __bf16 v16bf;
typedef __attribute__((ext_vector_type(8)))  __bf16 v8bf;
typedef __attribute__((ext_vector_type(8)))  float  v8f;
typedef __attribute__((ext_vector_type(4)))  float  v4f;

#define N_FEAT   16384            // K
#define EMB_DIM  64               // N
#define KTILE    32               // K per WMMA
#define KC       128              // K-chunk staged in LDS per buffer
#define RSTRIDE  (KC + 8)         // padded LDS row stride (elems): +16B kills
                                  // the 16-way bank conflict of a 256B stride
#define BUFELEM  (EMB_DIM * RSTRIDE)
#define NCHUNK   (N_FEAT / KC)    // 128
#define NKT      (KC / KTILE)     // 4 k-tiles per chunk

// ---------------------------------------------------------------------------
// Prep: E[k][n] (f32, row-major KxN) -> Bt[n][k] (bf16, row-major NxK)
// ---------------------------------------------------------------------------
__global__ void __launch_bounds__(256)
embed_transpose_bf16_kernel(const float* __restrict__ E,
                            __bf16* __restrict__ Bt) {
  int t = blockIdx.x * blockDim.x + threadIdx.x;   // 0 .. EMB_DIM*N_FEAT-1
  int n = t >> 14;                                 // t / N_FEAT
  int k = t & (N_FEAT - 1);
  Bt[(size_t)n * N_FEAT + k] = (__bf16)E[(size_t)k * EMB_DIM + n];
}

// Issue one async global->LDS 16-byte copy (gfx1250 VGLOBAL async op).
__device__ __forceinline__ void async_copy_b128(unsigned lds_byte_off,
                                                const __bf16* gsrc) {
  asm volatile("global_load_async_to_lds_b128 %0, %1, off"
               :
               : "v"(lds_byte_off),
                 "v"((unsigned long long)(uintptr_t)gsrc)
               : "memory");
}

__device__ __forceinline__ void wait_async_zero() {
  asm volatile("s_wait_asynccnt 0x0" ::: "memory");
}

// ---------------------------------------------------------------------------
// Main gather + GEMM kernel. 256 threads = 8 waves. Each wave: 16 rows x 64
// cols of output (4 accumulator tiles). Block covers 128 output rows.
// ---------------------------------------------------------------------------
__global__ void __launch_bounds__(256)
service_gather_gemm_kernel(const int* __restrict__ idx,
                           const float* __restrict__ S,     // [N_SRV][N_FEAT] f32
                           const __bf16* __restrict__ Bt,   // [EMB_DIM][N_FEAT] bf16
                           float* __restrict__ out) {       // [BATCH][EMB_DIM] f32
  extern __shared__ __align__(16) __bf16 sm[];              // 2 * BUFELEM elems

  const int tid   = threadIdx.x;
  const int lane  = tid & 31;
  const int wave  = tid >> 5;                      // 0..7
  const int m0    = (blockIdx.x * 8 + wave) * 16;  // first output row of wave
  const int mrow  = lane & 15;
  const int khalf = lane >> 4;                     // 0 or 1

  const unsigned smBase = (unsigned)(uintptr_t)sm; // LDS byte offset of base

  // --- cooperative async staging of one B K-chunk into LDS buffer `buf` ---
  // Chunk = Bt[0:64][kc : kc+KC], 64 rows x 256 B -> 1024 b128 segments,
  // 4 per thread, coalesced along K.
  auto stage_chunk = [&](int buf, int kc) {
#pragma unroll
    for (int i = 0; i < 4; ++i) {
      int s = i * 256 + tid;                       // 0..1023
      int n = s >> 4;                              // B row (0..63)
      int c = (s & 15) * 8;                        // elem offset in chunk row
      const __bf16* g = Bt + (size_t)n * N_FEAT + kc + c;
      unsigned l = smBase +
                   (unsigned)(buf * BUFELEM + n * RSTRIDE + c) * 2u;
      async_copy_b128(l, g);
    }
  };

  // Load the 4 B fragments (one per N-tile) for k-tile `kt` from LDS buffer.
  // B 32x16 lane layout: lanes 0-15 hold K {0..7, 16..23}, lanes 16-31 K {8..15, 24..31}.
  auto load_bfrags = [&](const __bf16* bbase, int kt,
                         v16bf& f0, v16bf& f1, v16bf& f2, v16bf& f3) {
    const int kcol = kt * KTILE + khalf * 8;
    const __bf16* bp0 = bbase + (mrow +  0) * RSTRIDE + kcol;
    const __bf16* bp1 = bbase + (mrow + 16) * RSTRIDE + kcol;
    const __bf16* bp2 = bbase + (mrow + 32) * RSTRIDE + kcol;
    const __bf16* bp3 = bbase + (mrow + 48) * RSTRIDE + kcol;
    v8bf p0 = *(const v8bf*)(bp0), q0 = *(const v8bf*)(bp0 + 16);
    v8bf p1 = *(const v8bf*)(bp1), q1 = *(const v8bf*)(bp1 + 16);
    v8bf p2 = *(const v8bf*)(bp2), q2 = *(const v8bf*)(bp2 + 16);
    v8bf p3 = *(const v8bf*)(bp3), q3 = *(const v8bf*)(bp3 + 16);
    f0 = __builtin_shufflevector(p0, q0, 0, 1, 2, 3, 4, 5, 6, 7,
                                 8, 9, 10, 11, 12, 13, 14, 15);
    f1 = __builtin_shufflevector(p1, q1, 0, 1, 2, 3, 4, 5, 6, 7,
                                 8, 9, 10, 11, 12, 13, 14, 15);
    f2 = __builtin_shufflevector(p2, q2, 0, 1, 2, 3, 4, 5, 6, 7,
                                 8, 9, 10, 11, 12, 13, 14, 15);
    f3 = __builtin_shufflevector(p3, q3, 0, 1, 2, 3, 4, 5, 6, 7,
                                 8, 9, 10, 11, 12, 13, 14, 15);
  };

  // Gathered A row for this lane's matrix row.
  const float* __restrict__ arow = S + (size_t)idx[m0 + mrow] * N_FEAT;
  // A 16x32 bf16 lane layout: lanes 0-15 hold K {0..7, 16..23},
  // lanes 16-31 hold K {8..15, 24..31} -> chunks at khalf*8 and +16.
  const int koffA = khalf * 8;

  v8f acc0 = {}, acc1 = {}, acc2 = {}, acc3 = {};

  // Prologue: stage chunk 0, preload A K-tile 0.
  stage_chunk(0, 0);
  v4f a00 = *(const v4f*)(arow + koffA);
  v4f a01 = *(const v4f*)(arow + koffA + 4);
  v4f a10 = *(const v4f*)(arow + koffA + 16);
  v4f a11 = *(const v4f*)(arow + koffA + 20);
  wait_async_zero();
  __syncthreads();

  for (int c = 0; c < NCHUNK; ++c) {
    // Kick off the next chunk's async staging while we compute this one.
    if (c + 1 < NCHUNK) stage_chunk((c + 1) & 1, (c + 1) * KC);

    const __bf16* __restrict__ bbase = sm + (size_t)(c & 1) * BUFELEM;

    // B fragment pipeline: issue k-tile 0's LDS loads now; tile kt+1's loads
    // are issued before tile kt's WMMAs execute.
    v16bf bc0, bc1, bc2, bc3;                      // current B fragments
    v16bf bn0, bn1, bn2, bn3;                      // next B fragments
    load_bfrags(bbase, 0, bc0, bc1, bc2, bc3);

#pragma unroll
    for (int kt = 0; kt < NKT; ++kt) {
      // ---- convert current A tile to bf16 fragment ----
      v16bf afrag;
#pragma unroll
      for (int i = 0; i < 4; ++i) {
        afrag[i]      = (__bf16)a00[i];
        afrag[4 + i]  = (__bf16)a01[i];
        afrag[8 + i]  = (__bf16)a10[i];
        afrag[12 + i] = (__bf16)a11[i];
      }

      // ---- issue next A tile's global loads (in flight during WMMAs) ----
      int kbn = c * KC + (kt + 1) * KTILE;
      if (kbn > N_FEAT - KTILE) kbn = N_FEAT - KTILE;   // clamp: never OOB
      const float* ap = arow + kbn + koffA;
      a00 = *(const v4f*)(ap);
      a01 = *(const v4f*)(ap + 4);
      a10 = *(const v4f*)(ap + 16);
      a11 = *(const v4f*)(ap + 20);

      // ---- issue next B tile's LDS loads (in flight during WMMAs) ----
      if (kt + 1 < NKT) load_bfrags(bbase, kt + 1, bn0, bn1, bn2, bn3);

      // ---- 4 WMMAs: D = A x B + C (f32 accumulate) ----
      acc0 = __builtin_amdgcn_wmma_f32_16x16x32_bf16(false, afrag, false, bc0,
                                                     (short)0, acc0, false, false);
      acc1 = __builtin_amdgcn_wmma_f32_16x16x32_bf16(false, afrag, false, bc1,
                                                     (short)0, acc1, false, false);
      acc2 = __builtin_amdgcn_wmma_f32_16x16x32_bf16(false, afrag, false, bc2,
                                                     (short)0, acc2, false, false);
      acc3 = __builtin_amdgcn_wmma_f32_16x16x32_bf16(false, afrag, false, bc3,
                                                     (short)0, acc3, false, false);

      bc0 = bn0; bc1 = bn1; bc2 = bn2; bc3 = bn3;
    }

    // Next buffer fully landed + everyone done reading before reuse.
    wait_async_zero();
    __syncthreads();
  }

  // ---- Store: C/D layout -> VGPR j holds row (khalf*8 + j), col = mrow ----
#pragma unroll
  for (int j = 0; j < 8; ++j) {
    float* __restrict__ o = out + (size_t)(m0 + khalf * 8 + j) * EMB_DIM + mrow;
    o[0]  = acc0[j];
    o[16] = acc1[j];
    o[32] = acc2[j];
    o[48] = acc3[j];
  }
}

// ---------------------------------------------------------------------------
// Host launcher. Inputs (setup_inputs order):
//   d_in[0] = data             int32  [16384]
//   d_in[1] = service_matrix   f32    [5825 * 16384]
//   d_in[2] = embedding_matrix f32    [16384 * 64]
// Output: f32 [16384 * 64].  d_ws holds the 2 MB bf16 transposed embedding.
// ---------------------------------------------------------------------------
extern "C" void kernel_launch(void* const* d_in, const int* in_sizes, int n_in,
                              void* d_out, int out_size, void* d_ws, size_t ws_size,
                              hipStream_t stream) {
  const int*   d_data = (const int*)d_in[0];
  const float* d_srv  = (const float*)d_in[1];
  const float* d_emb  = (const float*)d_in[2];
  float*       d_o    = (float*)d_out;
  __bf16*      d_bt   = (__bf16*)d_ws;            // 64 * 16384 * 2 B = 2 MB

  const int batch = in_sizes[0];                  // 16384

  // 1) Convert/transpose embedding matrix to bf16, N-major.
  {
    int total  = EMB_DIM * N_FEAT;                // 1,048,576 elements
    int blocks = total / 256;
    embed_transpose_bf16_kernel<<<blocks, 256, 0, stream>>>(d_emb, d_bt);
  }

  // 2) Gather + GEMM. 8 waves/block, 128 output rows per block.
  {
    int    blocks = batch / 128;                  // 128
    size_t shmem  = (size_t)2 * BUFELEM * sizeof(__bf16);  // ~34.8 KB
    service_gather_gemm_kernel<<<blocks, 256, shmem, stream>>>(d_data, d_srv,
                                                               d_bt, d_o);
  }
}
